// SwinResidualCrossAttention_51908974739556
// MI455X (gfx1250) — compile-verified, matrix operand order checked by reading, not run on hardware
//
#include <hip/hip_runtime.h>
#include <stdint.h>

// ---------------------------------------------------------------------------
// Swin-style residual cross window attention, fused, bf16 WMMA (gfx1250).
// B=8, H=W=128, C=256, window 8x8 (S=64), NH=8, HD=32.
// One workgroup per window (2048 workgroups), 256 threads = 8 wave32 = 1 wave
// per head. All matmuls use v_wmma_f32_16x16x32_bf16 (fp32 accumulate).
// ---------------------------------------------------------------------------

typedef __attribute__((ext_vector_type(16))) __bf16   v16bf;
typedef __attribute__((ext_vector_type(4)))  __bf16   v4bf;
typedef __attribute__((ext_vector_type(8)))  float    v8f;
typedef __attribute__((ext_vector_type(4)))  uint32_t v4u;

#define WMMA_BF16(a, b, c) \
  __builtin_amdgcn_wmma_f32_16x16x32_bf16(false, (a), false, (b), (short)0, (c), false, false)

constexpr int NW = 2048;   // number of windows

// LDS row strides (elements).  Both are multiples of 8 elements (16 B) so
// fragment chunks are b128-loadable, and both give conflict-free row->bank
// rotation: SX=280 -> 140 dw/row, 140%64=12 (r*12%64 distinct for r<16);
// SV=72 -> 36 dw/row, 36%64=36 (r*36%64 distinct for r<16).
constexpr int SX = 280;    // XN/RN/Q/K rows ([64][256] + pad)
constexpr int SV = 72;     // Vt rows ([256][64] + pad) and P rows

// LDS layout (bytes), all bases 16B-aligned.  Region A (XN+RN) is reused for
// softmax P in phase 3 (P = 8 heads * 64*72*2 = 73728 >= 2*35840).
// Q buffer is reused for the attention output (per-wave disjoint 32-col slabs).
constexpr int OFF_XN = 0;
constexpr int OFF_RN = 64 * SX * 2;                 // 35840
constexpr int OFF_P  = 0;                           // reuses XN+RN region
constexpr int REG_A  = 8 * 64 * SV * 2;             // 73728 (>= XN+RN)
constexpr int OFF_Q  = REG_A;                       // 73728
constexpr int OFF_K  = OFF_Q + 64 * SX * 2;         // 109568
constexpr int OFF_VT = OFF_K + 64 * SX * 2;         // 145408
constexpr int LDS_BYTES = OFF_VT + 256 * SV * 2;    // 182272  (< 320KB WGP budget)

// ---------------------------------------------------------------------------
// Row-major bf16 fragment loader.  Works for both A operand (r = M row) and
// B operand (r = N col), assuming B mirrors the A layout over N.
// ISA 16-bit A 16x32 layout: lanes 0-15 hold K=0..7 (v0-3) and K=16..23
// (v4-7); lanes 16-31 hold K=8..15 and K=24..31.  Each half is 16 contiguous
// bytes -> two b128 loads per lane (all bases/strides 16B-aligned).
// ---------------------------------------------------------------------------
__device__ __forceinline__ v16bf frag_ld(const __bf16* base, int stride, int lane) {
  const int r  = lane & 15;
  const int kb = (lane & 16) ? 8 : 0;
  const __bf16* p = base + r * stride + kb;
  union { v16bf v; v4u q[2]; } f;
  f.q[0] = *(const v4u*)(p);        // K: kb .. kb+7
  f.q[1] = *(const v4u*)(p + 16);   // K: 16+kb .. 16+kb+7
  return f.v;
}

// ---------------------------------------------------------------------------
// Weight pre-conversion: fp32 -> bf16 into workspace (L2-resident, 512 KB).
// d_ws layout: in_proj [768][256] bf16, then out_proj [256][256] bf16.
// ---------------------------------------------------------------------------
__global__ void cvt_wts(const float* __restrict__ inw, const float* __restrict__ outw,
                        __bf16* __restrict__ dst) {
  const int i = blockIdx.x * blockDim.x + threadIdx.x;
  if (i < 768 * 256) dst[i] = (__bf16)inw[i];
  if (i < 256 * 256) dst[768 * 256 + i] = (__bf16)outw[i];
}

// ---------------------------------------------------------------------------
// Main fused kernel.
// ---------------------------------------------------------------------------
__global__ __launch_bounds__(256) void swin_attn_kernel(
    const float* __restrict__ x,   const float* __restrict__ res,
    const float* __restrict__ lxw, const float* __restrict__ lxb,
    const float* __restrict__ lrw, const float* __restrict__ lrb,
    const float* __restrict__ ipb, const float* __restrict__ opb,
    const __bf16* __restrict__ wbf, float* __restrict__ out) {
  extern __shared__ __align__(16) char smem[];
  __bf16* XN = (__bf16*)(smem + OFF_XN);
  __bf16* RN = (__bf16*)(smem + OFF_RN);
  __bf16* Pb = (__bf16*)(smem + OFF_P);
  __bf16* Qb = (__bf16*)(smem + OFF_Q);   // Q, later reused as attention output
  __bf16* Kb = (__bf16*)(smem + OFF_K);
  __bf16* Vt = (__bf16*)(smem + OFF_VT);  // V transposed: [d(0..255)][s(0..63)]

  const int wid  = (int)blockIdx.x;
  const int b    = wid >> 8;              // / (16*16)
  const int wh   = (wid >> 4) & 15;
  const int ww   = wid & 15;
  const int tid  = (int)threadIdx.x;
  const int lane = tid & 31;
  const int wave = tid >> 5;
  const int nl15 = lane & 15;
  const int mhi  = (lane & 16) ? 8 : 0;

  // ---- Phase 1: LayerNorm(x)->XN bf16, LayerNorm(residual)->RN bf16 ------
  {
    const int s  = tid >> 2;      // token 0..63, 4 threads per token
    const int qd = tid & 3;       // channel quarter
    const int rh = s >> 3, rw = s & 7;
    const size_t tok = ((size_t)((b * 128 + wh * 8 + rh) * 128 + ww * 8 + rw)) * 256;
    for (int which = 0; which < 2; ++which) {
      const float* src = which ? res : x;
      const float* gw  = which ? lrw : lxw;
      const float* gb  = which ? lrb : lxb;
      __bf16* dst = which ? RN : XN;
      const float4* p = (const float4*)(src + tok + (size_t)qd * 64);
      float4 vb[16];
      float sum = 0.f, sq = 0.f;
#pragma unroll
      for (int i = 0; i < 16; ++i) {
        vb[i] = p[i];
        sum += vb[i].x + vb[i].y + vb[i].z + vb[i].w;
        sq  += vb[i].x * vb[i].x + vb[i].y * vb[i].y + vb[i].z * vb[i].z + vb[i].w * vb[i].w;
      }
      // reduce over the 4 lanes covering this token (aligned group of 4)
      sum += __shfl_xor(sum, 1); sum += __shfl_xor(sum, 2);
      sq  += __shfl_xor(sq, 1);  sq  += __shfl_xor(sq, 2);
      const float mean = sum * (1.0f / 256.0f);
      const float var  = sq * (1.0f / 256.0f) - mean * mean;
      const float rstd = rsqrtf(var + 1e-5f);
#pragma unroll
      for (int i = 0; i < 16; ++i) {
        const int c = qd * 64 + i * 4;
        v4bf o;                       // 4 consecutive channels -> one 8B store
        o[0] = (__bf16)((vb[i].x - mean) * rstd * gw[c + 0] + gb[c + 0]);
        o[1] = (__bf16)((vb[i].y - mean) * rstd * gw[c + 1] + gb[c + 1]);
        o[2] = (__bf16)((vb[i].z - mean) * rstd * gw[c + 2] + gb[c + 2]);
        o[3] = (__bf16)((vb[i].w - mean) * rstd * gw[c + 3] + gb[c + 3]);
        *(v4bf*)(dst + s * SX + c) = o;
      }
    }
  }
  __syncthreads();

  // ---- Phase 2: Q = XN Wq^T + bq, K = RN Wk^T + bk, V = RN Wv^T + bv -----
  // Wave w owns output columns [32w, 32w+32).  V stored transposed into Vt.
  // mat loop fully unrolled so the store destination is compile-time constant.
#pragma unroll
  for (int mat = 0; mat < 3; ++mat) {
    const __bf16* src  = (mat == 0) ? XN : RN;
    const __bf16* wrow = wbf + mat * 256 * 256;     // [n][k] row-major bf16
    for (int nj = 0; nj < 2; ++nj) {
      const int n0 = wave * 32 + nj * 16;
      const float bn = ipb[mat * 256 + n0 + nl15];
      for (int mi = 0; mi < 4; ++mi) {
        v8f acc = {0.f, 0.f, 0.f, 0.f, 0.f, 0.f, 0.f, 0.f};
#pragma unroll
        for (int kt = 0; kt < 8; ++kt) {
          v16bf a  = frag_ld(src + mi * 16 * SX + kt * 32, SX, lane);
          v16bf bb = frag_ld(wrow + n0 * 256 + kt * 32, 256, lane);
          acc = WMMA_BF16(a, bb, acc);
        }
        const int nc = n0 + nl15;
        const int mb = mi * 16 + mhi;
#pragma unroll
        for (int i = 0; i < 8; ++i) {
          const float v = acc[i] + bn;           // C/D layout: vgpr i -> M=mb+i, lane -> N=nc
          if (mat == 0)      Qb[(mb + i) * SX + nc] = (__bf16)v;
          else if (mat == 1) Kb[(mb + i) * SX + nc] = (__bf16)v;
          else               Vt[nc * SV + (mb + i)] = (__bf16)v;   // transposed
        }
      }
    }
  }
  __syncthreads();

  // ---- Phase 3: per-head attention (wave = head).  HD=32 -> 1 K-step. ----
  // One 16-row block (mi) at a time keeps only 4 score tiles (32 fp32 VGPRs)
  // live through the softmax -> stays under 256 VGPRs.
  {
    const int h = wave;
    __bf16* Ph = Pb + h * (64 * SV);
    const float scale = 0.17677669529663687f;    // 1/sqrt(32)
    for (int mi = 0; mi < 4; ++mi) {
      v8f sc[4];
      v16bf a = frag_ld(Qb + mi * 16 * SX + h * 32, SX, lane);
#pragma unroll
      for (int ni = 0; ni < 4; ++ni) {
        v8f z = {0.f, 0.f, 0.f, 0.f, 0.f, 0.f, 0.f, 0.f};
        v16bf bb = frag_ld(Kb + ni * 16 * SX + h * 32, SX, lane);
        sc[ni] = WMMA_BF16(a, bb, z);
      }
      // softmax over N (64 cols): per-VGPR row, reduce across the 16-lane half
#pragma unroll
      for (int i = 0; i < 8; ++i) {
        float mx = -3.0e38f;
#pragma unroll
        for (int ni = 0; ni < 4; ++ni) mx = fmaxf(mx, sc[ni][i] * scale);
        mx = fmaxf(mx, __shfl_xor(mx, 1)); mx = fmaxf(mx, __shfl_xor(mx, 2));
        mx = fmaxf(mx, __shfl_xor(mx, 4)); mx = fmaxf(mx, __shfl_xor(mx, 8));
        float e[4], se = 0.f;
#pragma unroll
        for (int ni = 0; ni < 4; ++ni) { e[ni] = __expf(sc[ni][i] * scale - mx); se += e[ni]; }
        se += __shfl_xor(se, 1); se += __shfl_xor(se, 2);
        se += __shfl_xor(se, 4); se += __shfl_xor(se, 8);
        const float inv = 1.0f / se;
        const int m = mi * 16 + mhi + i;
#pragma unroll
        for (int ni = 0; ni < 4; ++ni)
          Ph[m * SV + ni * 16 + nl15] = (__bf16)(e[ni] * inv);
      }
    }
    // out_h = P @ V_h  -> overwrite Qb columns [32h, 32h+32) (wave-private)
    for (int dj = 0; dj < 2; ++dj) {
      const int n0 = h * 32 + dj * 16;
      for (int mi = 0; mi < 4; ++mi) {
        v8f acc = {0.f, 0.f, 0.f, 0.f, 0.f, 0.f, 0.f, 0.f};
#pragma unroll
        for (int kt = 0; kt < 2; ++kt) {
          v16bf a  = frag_ld(Ph + mi * 16 * SV + kt * 32, SV, lane);
          v16bf bb = frag_ld(Vt + n0 * SV + kt * 32, SV, lane);
          acc = WMMA_BF16(a, bb, acc);
        }
        const int nc = n0 + nl15;
        const int mb = mi * 16 + mhi;
#pragma unroll
        for (int i = 0; i < 8; ++i)
          Qb[(mb + i) * SX + nc] = (__bf16)acc[i];
      }
    }
  }
  __syncthreads();

  // ---- Phase 4: out = OUT Wo^T + bo, scattered to un-windowed layout -----
  {
    const __bf16* wo = wbf + 768 * 256;   // [n][k] bf16
    for (int nj = 0; nj < 2; ++nj) {
      const int n0 = wave * 32 + nj * 16;
      const float bn = opb[n0 + nl15];
      for (int mi = 0; mi < 4; ++mi) {
        v8f acc = {0.f, 0.f, 0.f, 0.f, 0.f, 0.f, 0.f, 0.f};
#pragma unroll
        for (int kt = 0; kt < 8; ++kt) {
          v16bf a  = frag_ld(Qb + mi * 16 * SX + kt * 32, SX, lane);
          v16bf bb = frag_ld(wo + n0 * 256 + kt * 32, 256, lane);
          acc = WMMA_BF16(a, bb, acc);
        }
        const int nc = n0 + nl15;
        const int mb = mi * 16 + mhi;
#pragma unroll
        for (int i = 0; i < 8; ++i) {
          const int s  = mb + i;
          const int rh = s >> 3, rw = s & 7;
          const size_t off =
              ((size_t)((b * 128 + wh * 8 + rh) * 128 + ww * 8 + rw)) * 256 + nc;
          out[off] = acc[i] + bn;   // lanes 0-15: 64B contiguous per row
        }
      }
    }
  }
}

// ---------------------------------------------------------------------------
extern "C" void kernel_launch(void* const* d_in, const int* in_sizes, int n_in,
                              void* d_out, int out_size, void* d_ws, size_t ws_size,
                              hipStream_t stream) {
  (void)in_sizes; (void)n_in; (void)out_size; (void)ws_size;
  const float* x   = (const float*)d_in[0];
  const float* res = (const float*)d_in[1];
  const float* lxw = (const float*)d_in[2];
  const float* lxb = (const float*)d_in[3];
  const float* lrw = (const float*)d_in[4];
  const float* lrb = (const float*)d_in[5];
  const float* ipw = (const float*)d_in[6];
  const float* ipb = (const float*)d_in[7];
  const float* opw = (const float*)d_in[8];
  const float* opb = (const float*)d_in[9];
  float* out  = (float*)d_out;
  __bf16* wbf = (__bf16*)d_ws;   // needs 512 KB of workspace

  // 1) weight fp32 -> bf16 (deterministic each call; weights stay in L2)
  cvt_wts<<<(768 * 256 + 255) / 256, 256, 0, stream>>>(ipw, opw, wbf);

  // 2) fused window-attention kernel, 182,272 B dynamic LDS per workgroup
  hipFuncSetAttribute(reinterpret_cast<const void*>(swin_attn_kernel),
                      hipFuncAttributeMaxDynamicSharedMemorySize, LDS_BYTES);
  swin_attn_kernel<<<NW, 256, LDS_BYTES, stream>>>(x, res, lxw, lxb, lrw, lrb,
                                                   ipb, opb, wbf, out);
}
